// GroupedQueryAttention_10900626997689
// MI455X (gfx1250) — compile-verified, matrix operand order checked by reading, not run on hardware
//
#include <hip/hip_runtime.h>
#include <hip/hip_bf16.h>

// ---------------------------------------------------------------------------
// GQA attention for MI455X (gfx1250, wave32).
// Heavy math: v_wmma_f32_16x16x32_bf16 (f32 accumulate).
// Tile staging: Tensor Data Mover (tensor_load_to_lds) where available.
// ---------------------------------------------------------------------------

typedef __bf16          v8bf  __attribute__((ext_vector_type(8)));
typedef __bf16          v16bf __attribute__((ext_vector_type(16)));
typedef float           v8f   __attribute__((ext_vector_type(8)));
typedef unsigned short  v8us  __attribute__((ext_vector_type(8)));
typedef unsigned int    v4u   __attribute__((ext_vector_type(4)));
typedef int             v8i   __attribute__((ext_vector_type(8)));
typedef int             v4i   __attribute__((ext_vector_type(4)));

#if defined(__has_builtin)
#if __has_builtin(__builtin_amdgcn_tensor_load_to_lds) && \
    __has_builtin(__builtin_amdgcn_s_wait_tensorcnt)
#define HAVE_TDM 1
#endif
#endif
#ifndef HAVE_TDM
#define HAVE_TDM 0
#endif

#define D_MODEL      2048
#define NUM_Q_HEADS  32
#define NUM_KV_HEADS 8
#define HEAD_DIM     64
#define NUM_GROUPS   4
#define KV_DIM       (NUM_KV_HEADS * HEAD_DIM)   // 512
#define BATCH        2
#define SEQ          2048
#define MROWS        (BATCH * SEQ)               // 4096

// round-to-nearest-even f32 -> bf16 bits
static __device__ __forceinline__ unsigned short f2bf(float f) {
  unsigned int u = __builtin_bit_cast(unsigned int, f);
  u += 0x7FFFu + ((u >> 16) & 1u);
  return (unsigned short)(u >> 16);
}

// Build a 16-element bf16 fragment from two 16-byte chunks (ds/global b128).
static __device__ __forceinline__ v16bf ldfrag(const unsigned short* lo,
                                               const unsigned short* hi) {
  v8bf a = *(const v8bf*)lo;
  v8bf b = *(const v8bf*)hi;
  return __builtin_shufflevector(a, b, 0,1,2,3,4,5,6,7,8,9,10,11,12,13,14,15);
}

static __device__ __forceinline__ v8f wmma_bf16(v16bf a, v16bf b, v8f c) {
  // (neg_a, A, neg_b, B, c_mod, C, reuse_a, reuse_b)
  return __builtin_amdgcn_wmma_f32_16x16x32_bf16(false, a, false, b,
                                                 (short)0, c, false, false);
}

static __device__ __forceinline__ float redmax16(float v) {
  v = fmaxf(v, __shfl_xor(v, 1, 32));
  v = fmaxf(v, __shfl_xor(v, 2, 32));
  v = fmaxf(v, __shfl_xor(v, 4, 32));
  v = fmaxf(v, __shfl_xor(v, 8, 32));
  return v;   // reduced within 16-lane halves (rows r / r+8 stay separate)
}
static __device__ __forceinline__ float redsum16(float v) {
  v += __shfl_xor(v, 1, 32);
  v += __shfl_xor(v, 2, 32);
  v += __shfl_xor(v, 4, 32);
  v += __shfl_xor(v, 8, 32);
  return v;
}

#if HAVE_TDM
// ---------------------------------------------------------------------------
// TDM: load a rows x cols bf16 tile (row stride src_stride elems) from global
// into LDS at lds_byte, inserting 16B of LDS padding after every 128B row
// (pad_interval=32 DWORDs -> code 4, pad_amount=4 DWORDs -> code 3), matching
// the software (cols=64, LDS row stride=144B) tile layout used below.
// Descriptor bit layout per CDNA5 ISA sec 8.3-8.5. Wave-uniform args only;
// issue from one wave, then s_wait_tensorcnt(0) before the workgroup barrier.
// ---------------------------------------------------------------------------
static __device__ __forceinline__ void tdm_load_2d_bf16(
    const unsigned short* gsrc, unsigned int lds_byte,
    int rows, int cols, long long src_stride_elems) {
  unsigned long long ga = (unsigned long long)gsrc;
  v4u g0;
  g0[0] = 1u;                                        // count=1, user mode
  g0[1] = lds_byte;                                  // lds_addr (bytes)
  g0[2] = (unsigned int)(ga & 0xFFFFFFFFu);          // global_addr[31:0]
  g0[3] = (unsigned int)((ga >> 32) & 0x01FFFFFFu)   // global_addr[56:32]
        | (2u << 30);                                // type = 2 ("image")
  const unsigned int td = 0x40000000u;               // huge dims: tiles in-bounds
  v8i g1;
  g1[0] = (int)((1u << 16)        // data_size = 1 -> 2 bytes/elem
        | (1u << 20)              // pad_enable
        | (4u << 22)              // pad_interval: 32 DWORDs (128B)
        | (3u << 25));            // pad_amount:   4 DWORDs (16B)
  g1[1] = (int)((td & 0xFFFFu) << 16);                       // tensor_dim0 lo
  g1[2] = (int)(((td >> 16) & 0xFFFFu) | ((td & 0xFFFFu) << 16)); // d0 hi | d1 lo
  g1[3] = (int)(((td >> 16) & 0xFFFFu)
        | (((unsigned int)cols & 0xFFFFu) << 16));           // tile_dim0
  g1[4] = (int)((unsigned int)rows & 0xFFFFu);               // tile_dim1 (dim2=0)
  unsigned long long s0 = (unsigned long long)src_stride_elems;
  g1[5] = (int)(s0 & 0xFFFFFFFFu);                           // dim0_stride lo
  g1[6] = (int)((s0 >> 32) & 0xFFFFu);                       // dim0_stride hi
  g1[7] = 0;
  v4i z4 = {0, 0, 0, 0};                                     // 2D: groups 2/3 unused
  v8i z8 = {0, 0, 0, 0, 0, 0, 0, 0};
  __builtin_amdgcn_tensor_load_to_lds(g0, g1, z4, z4, z8, 0);
}
#endif

// ---------------------------------------------------------------------------
// f32 -> bf16 conversion (vectorized 4-wide)
// ---------------------------------------------------------------------------
__global__ void cvt_f32_bf16(const float* __restrict__ in,
                             unsigned short* __restrict__ out, int n) {
  int i = (blockIdx.x * blockDim.x + threadIdx.x) * 4;
  if (i + 3 < n) {
    float4 v = *(const float4*)(in + i);
    ushort4 o;
    o.x = f2bf(v.x); o.y = f2bf(v.y); o.z = f2bf(v.z); o.w = f2bf(v.w);
    *(ushort4*)(out + i) = o;
  }
}

// ---------------------------------------------------------------------------
// Generic bf16 GEMM: C[M,N] = A[M,K] @ B[K,N], f32 accumulate.
// Block tile 128x128x64, 256 threads = 8 waves (2x4), wave tile 64x32.
// A tile staged by the TDM; B tile staged transposed by the vector pipes.
// ---------------------------------------------------------------------------
#define BM 128
#define BN 128
#define BK 64
#define LDK (BK + 8)   // pad: 144B row stride, 16B aligned

template <bool OUT_BF16>
__global__ __launch_bounds__(256)
void gemm_bf16_kernel(const unsigned short* __restrict__ A,
                      const unsigned short* __restrict__ Bm,
                      float* __restrict__ Cf,
                      unsigned short* __restrict__ Cb,
                      int M, int N, int K) {
  __shared__ unsigned short As[BM][LDK];
  __shared__ unsigned short Bs[BN][LDK];   // stored transposed: [n][k]

  const int t    = threadIdx.x;
  const int lane = t & 31;
  const int wave = t >> 5;
  const int wm   = wave & 1;    // 2 waves along M
  const int wn   = wave >> 1;   // 4 waves along N
  const int l15  = lane & 15;
  const int kh   = lane >> 4;   // which K-half this lane holds
  const int m0   = blockIdx.y * BM;
  const int n0   = blockIdx.x * BN;

  v8f acc[4][2];
#pragma unroll
  for (int i = 0; i < 4; ++i)
#pragma unroll
    for (int j = 0; j < 2; ++j) acc[i][j] = {};

  for (int k0 = 0; k0 < K; k0 += BK) {
#if HAVE_TDM
    // --- A tile 128x64 via Tensor Data Mover (wave 0 issues, EXEC ignored) ---
    if (wave == 0) {
      tdm_load_2d_bf16(A + (size_t)m0 * K + k0,
                       (unsigned int)(size_t)(void*)As,
                       BM, BK, (long long)K);
    }
#else
#pragma unroll
    for (int i = 0; i < 4; ++i) {
      int idx = t + i * 256;
      int row = idx >> 3, cg = (idx & 7) * 8;
      v8bf v = *(const v8bf*)(A + (size_t)(m0 + row) * K + k0 + cg);
      *(v8bf*)&As[row][cg] = v;
    }
#endif
    // --- stage B tile transposed: global [k][n] -> LDS [n][k] ---
#pragma unroll
    for (int i = 0; i < 4; ++i) {
      int idx = t + i * 256;
      int kr = idx >> 4, ng = (idx & 15) * 8;
      v8bf v = *(const v8bf*)(Bm + (size_t)(k0 + kr) * N + n0 + ng);
      v8us u = __builtin_bit_cast(v8us, v);
#pragma unroll
      for (int j = 0; j < 8; ++j) Bs[ng + j][kr] = u[j];
    }
    // hint the next K tile toward the caches while we compute this one
    if (k0 + BK < K) {
      __builtin_prefetch(A + (size_t)(m0 + (t >> 1)) * K + k0 + BK, 0, 0);
      __builtin_prefetch(Bm + (size_t)(k0 + BK + (t >> 3)) * N + n0, 0, 0);
    }
#if HAVE_TDM
    if (wave == 0) __builtin_amdgcn_s_wait_tensorcnt(0);
#endif
    __syncthreads();

#pragma unroll
    for (int s = 0; s < 2; ++s) {        // two K=32 WMMA steps per tile
      const int ks = s * 32;
      v16bf af[4], bfv[2];
#pragma unroll
      for (int i = 0; i < 4; ++i) {
        int row = wm * 64 + i * 16 + l15;
        af[i] = ldfrag(&As[row][ks + kh * 8], &As[row][ks + 16 + kh * 8]);
      }
#pragma unroll
      for (int j = 0; j < 2; ++j) {
        int col = wn * 32 + j * 16 + l15;
        bfv[j] = ldfrag(&Bs[col][ks + kh * 8], &Bs[col][ks + 16 + kh * 8]);
      }
#pragma unroll
      for (int i = 0; i < 4; ++i)
#pragma unroll
        for (int j = 0; j < 2; ++j)
          acc[i][j] = wmma_bf16(af[i], bfv[j], acc[i][j]);
    }
    __syncthreads();
  }

  // epilogue: C layout — VGPR r holds row r (lanes 0-15) / r+8 (lanes 16-31)
#pragma unroll
  for (int i = 0; i < 4; ++i)
#pragma unroll
    for (int j = 0; j < 2; ++j)
#pragma unroll
      for (int r = 0; r < 8; ++r) {
        int row = m0 + wm * 64 + i * 16 + r + kh * 8;
        int col = n0 + wn * 32 + j * 16 + l15;
        float v = acc[i][j][r];
        if (OUT_BF16) Cb[(size_t)row * N + col] = f2bf(v);
        else          Cf[(size_t)row * N + col] = v;
      }
}

// ---------------------------------------------------------------------------
// Flash attention, GQA. One block = (batch b, q-head h, 64-query tile).
// 128 threads = 4 waves; each wave owns 16 queries. Online softmax.
// K tile staged by the TDM; V tile transposed by the vector pipes.
// ---------------------------------------------------------------------------
#define KT 64            // key tile
#define LDF 72           // padded LDS row (144B, 16B aligned)

__global__ __launch_bounds__(128)
void flash_gqa_kernel(const unsigned short* __restrict__ Q,
                      const unsigned short* __restrict__ Kb,
                      const unsigned short* __restrict__ Vb,
                      unsigned short* __restrict__ O) {
  __shared__ unsigned short Ks [KT][LDF];        // [key][d]
  __shared__ unsigned short Vts[HEAD_DIM][LDF];  // transposed: [d][key]
  __shared__ unsigned short Ps [4][16][LDF];     // per-wave P: [q][key]

  const int t    = threadIdx.x;
  const int lane = t & 31;
  const int wave = t >> 5;
  const int l15  = lane & 15;
  const int kh   = lane >> 4;
  const int b    = blockIdx.z;
  const int h    = blockIdx.y;
  const int kvh  = h / NUM_GROUPS;               // repeat_interleave mapping
  const int q0   = blockIdx.x * 64;
  const int qb   = q0 + wave * 16;               // this wave's 16 queries
  const float inv_scale = 0.125f;                // 1/sqrt(64)

  unsigned short (*Pw)[LDF] = Ps[wave];

  // Q fragments for this wave: A operand [16 q x 64 d] -> two K=32 frags
  v16bf qa[2];
#pragma unroll
  for (int f = 0; f < 2; ++f) {
    size_t base = (size_t)(b * SEQ + qb + l15) * D_MODEL + h * HEAD_DIM;
    qa[f] = ldfrag(Q + base + f * 32 + kh * 8,
                   Q + base + f * 32 + 16 + kh * 8);
  }

  v8f Oacc[4];
#pragma unroll
  for (int dt = 0; dt < 4; ++dt) Oacc[dt] = {};
  float m_run[8], l_run[8];
#pragma unroll
  for (int r = 0; r < 8; ++r) { m_run[r] = -3.0e38f; l_run[r] = 0.0f; }

  for (int kt = 0; kt < SEQ / KT; ++kt) {
    const size_t tile_base =
        (size_t)(b * SEQ + kt * KT) * KV_DIM + kvh * HEAD_DIM;
#if HAVE_TDM
    // --- K tile 64x64 via Tensor Data Mover ---
    if (wave == 0) {
      tdm_load_2d_bf16(Kb + tile_base,
                       (unsigned int)(size_t)(void*)Ks,
                       KT, HEAD_DIM, (long long)KV_DIM);
    }
#endif
    // --- stage V tile transposed [d][key] (and K when no TDM) ---
#pragma unroll
    for (int i = 0; i < 4; ++i) {
      int idx = t + i * 128;
      int row = idx >> 3, cg = (idx & 7) * 8;
      size_t src = tile_base + (size_t)row * KV_DIM + cg;
#if !HAVE_TDM
      v8bf kv = *(const v8bf*)(Kb + src);
      *(v8bf*)&Ks[row][cg] = kv;
#endif
      v8bf vv = *(const v8bf*)(Vb + src);
      v8us u = __builtin_bit_cast(v8us, vv);
#pragma unroll
      for (int j = 0; j < 8; ++j) Vts[cg + j][row] = u[j];
    }
#if HAVE_TDM
    if (wave == 0) __builtin_amdgcn_s_wait_tensorcnt(0);
#endif
    __syncthreads();

    // --- scores: S = (Q @ K^T) * inv_scale, tile 16 q x 64 key ---
    v8f sacc[4];
#pragma unroll
    for (int nt = 0; nt < 4; ++nt) sacc[nt] = {};
#pragma unroll
    for (int nt = 0; nt < 4; ++nt) {
      int kcol = nt * 16 + l15;   // B-operand lane = key column
#pragma unroll
      for (int f = 0; f < 2; ++f) {
        v16bf kb = ldfrag(&Ks[kcol][f * 32 + kh * 8],
                          &Ks[kcol][f * 32 + 16 + kh * 8]);
        sacc[nt] = wmma_bf16(qa[f], kb, sacc[nt]);
      }
    }
#pragma unroll
    for (int nt = 0; nt < 4; ++nt)
#pragma unroll
      for (int r = 0; r < 8; ++r) sacc[nt][r] *= inv_scale;

    // --- online softmax per row (16-lane reductions) ---
#pragma unroll
    for (int r = 0; r < 8; ++r) {
      float rmax = fmaxf(fmaxf(sacc[0][r], sacc[1][r]),
                         fmaxf(sacc[2][r], sacc[3][r]));
      rmax = redmax16(rmax);
      float mnew = fmaxf(m_run[r], rmax);
      float corr = __expf(m_run[r] - mnew);
      m_run[r] = mnew;
      float psum = 0.0f;
      int prow = r + kh * 8;
#pragma unroll
      for (int nt = 0; nt < 4; ++nt) {
        float p = __expf(sacc[nt][r] - mnew);
        psum += p;
        Pw[prow][nt * 16 + l15] = f2bf(p);
      }
      psum = redsum16(psum);
      l_run[r] = l_run[r] * corr + psum;
#pragma unroll
      for (int dt = 0; dt < 4; ++dt) Oacc[dt][r] *= corr;
    }

    // --- O += P @ V  (P from per-wave LDS as A operand) ---
#pragma unroll
    for (int kf = 0; kf < 2; ++kf) {
      v16bf pa = ldfrag(&Pw[l15][kf * 32 + kh * 8],
                        &Pw[l15][kf * 32 + 16 + kh * 8]);
#pragma unroll
      for (int dt = 0; dt < 4; ++dt) {
        int dd = dt * 16 + l15;
        v16bf vb = ldfrag(&Vts[dd][kf * 32 + kh * 8],
                          &Vts[dd][kf * 32 + 16 + kh * 8]);
        Oacc[dt] = wmma_bf16(pa, vb, Oacc[dt]);
      }
    }
    __syncthreads();
  }

  // --- epilogue: O / l -> bf16 attn-out [B*S, D_MODEL] ---
#pragma unroll
  for (int dt = 0; dt < 4; ++dt)
#pragma unroll
    for (int r = 0; r < 8; ++r) {
      int q = qb + r + kh * 8;
      int dd = dt * 16 + l15;
      float v = Oacc[dt][r] / l_run[r];
      O[(size_t)(b * SEQ + q) * D_MODEL + h * HEAD_DIM + dd] = f2bf(v);
    }
}

// ---------------------------------------------------------------------------
// Host launcher
// ---------------------------------------------------------------------------
extern "C" void kernel_launch(void* const* d_in, const int* in_sizes, int n_in,
                              void* d_out, int out_size, void* d_ws, size_t ws_size,
                              hipStream_t stream) {
  const float* x  = (const float*)d_in[0];
  const float* Wq = (const float*)d_in[1];
  const float* Wk = (const float*)d_in[2];
  const float* Wv = (const float*)d_in[3];
  const float* Wo = (const float*)d_in[4];
  float* out = (float*)d_out;

  char* ws = (char*)d_ws;
  size_t off = 0;
  auto alloc = [&](size_t elems) {
    unsigned short* p = (unsigned short*)(ws + off);
    off += ((elems * sizeof(unsigned short)) + 255) & ~(size_t)255;
    return p;
  };
  unsigned short* xb  = alloc((size_t)MROWS * D_MODEL);
  unsigned short* wqb = alloc((size_t)D_MODEL * D_MODEL);
  unsigned short* wkb = alloc((size_t)D_MODEL * KV_DIM);
  unsigned short* wvb = alloc((size_t)D_MODEL * KV_DIM);
  unsigned short* wob = alloc((size_t)D_MODEL * D_MODEL);
  unsigned short* Qb  = alloc((size_t)MROWS * D_MODEL);
  unsigned short* Kb  = alloc((size_t)MROWS * KV_DIM);
  unsigned short* Vb  = alloc((size_t)MROWS * KV_DIM);
  unsigned short* Ob  = alloc((size_t)MROWS * D_MODEL);

  auto cvt = [&](const float* src, unsigned short* dst, size_t n) {
    int blocks = (int)((n / 4 + 255) / 256);
    cvt_f32_bf16<<<blocks, 256, 0, stream>>>(src, dst, (int)n);
  };
  cvt(x,  xb,  (size_t)MROWS * D_MODEL);
  cvt(Wq, wqb, (size_t)D_MODEL * D_MODEL);
  cvt(Wk, wkb, (size_t)D_MODEL * KV_DIM);
  cvt(Wv, wvb, (size_t)D_MODEL * KV_DIM);
  cvt(Wo, wob, (size_t)D_MODEL * D_MODEL);

  dim3 blk(256);
  // Q/K/V projections (bf16 outputs)
  gemm_bf16_kernel<true><<<dim3(D_MODEL / BN, MROWS / BM), blk, 0, stream>>>(
      xb, wqb, nullptr, Qb, MROWS, D_MODEL, D_MODEL);
  gemm_bf16_kernel<true><<<dim3(KV_DIM / BN, MROWS / BM), blk, 0, stream>>>(
      xb, wkb, nullptr, Kb, MROWS, KV_DIM, D_MODEL);
  gemm_bf16_kernel<true><<<dim3(KV_DIM / BN, MROWS / BM), blk, 0, stream>>>(
      xb, wvb, nullptr, Vb, MROWS, KV_DIM, D_MODEL);

  // flash attention
  flash_gqa_kernel<<<dim3(SEQ / 64, NUM_Q_HEADS, BATCH), dim3(128), 0, stream>>>(
      Qb, Kb, Vb, Ob);

  // output projection (f32 -> d_out)
  gemm_bf16_kernel<false><<<dim3(D_MODEL / BN, MROWS / BM), blk, 0, stream>>>(
      Ob, wob, out, nullptr, MROWS, D_MODEL, D_MODEL);
}
